// Attention_53678501266130
// MI455X (gfx1250) — matrix operand order repaired and checked by reading, not compiled, not run on hardware
//
#include <hip/hip_runtime.h>
#include <hip/hip_bf16.h>
#include <math.h>

// ---------------------------------------------------------------------------
// Attention block for MI455X (gfx1250, wave32, WMMA).
//   x:[4,256,64,64] f32 -> groupnorm -> qkv (bf16 WMMA) -> flash attention
//   (bf16 WMMA, online softmax, fp32 accum, async double-buffered LDS)
//   -> out proj + residual (f32 out)
// ---------------------------------------------------------------------------

typedef __attribute__((ext_vector_type(16))) __bf16 v16bf;
typedef __attribute__((ext_vector_type(8)))  __bf16 v8bf;
typedef __attribute__((ext_vector_type(8)))  float  v8f;
typedef __attribute__((ext_vector_type(4)))  int    v4i;

typedef __attribute__((address_space(1))) v4i g_v4i;   // global int4
typedef __attribute__((address_space(3))) v4i l_v4i;   // LDS int4

#if defined(__has_builtin)
#  if __has_builtin(__builtin_amdgcn_global_load_async_to_lds_b128) && \
      __has_builtin(__builtin_amdgcn_s_wait_asynccnt)
#    define USE_ASYNC_LDS 1
#  endif
#endif

#define BATCH  4
#define CH     256
#define NTOK   4096            // 64*64
#define NC     (NTOK * CH)
#define GROUPS 32
#define CPG    (CH / GROUPS)   // 8 channels per group

// ---- WMMA fragment loaders (both operands K-contiguous row-major) ---------
// A fragment: M = lane&15, lane<16 holds K=0..7 & 16..23, lane>=16 K=8..15 & 24..31
static __device__ __forceinline__ v16bf frag_a_rowmajor(const __bf16* base, int row0,
                                                        int ld, int koff) {
  const int lane = threadIdx.x & 31;
  const int r = lane & 15, h = lane >> 4;
  const __bf16* p = base + (size_t)(row0 + r) * ld + koff + h * 8;
  v8bf lo = *(const v8bf*)p;
  v8bf hi = *(const v8bf*)(p + 16);
  v16bf f;
#pragma unroll
  for (int i = 0; i < 8; ++i) { f[i] = lo[i]; f[i + 8] = hi[i]; }
  return f;
}

// B fragment: N = lane&15, lane<16 holds K=0..7 & 16..23, lane>=16 K=8..15 & 24..31
static __device__ __forceinline__ v16bf frag_b_rowmajor(const __bf16* base, int col0,
                                                        int ld, int koff) {
  const int lane = threadIdx.x & 31;
  const int c = lane & 15, h = lane >> 4;
  const __bf16* p = base + (size_t)(col0 + c) * ld + koff + h * 8;
  v8bf lo = *(const v8bf*)p;
  v8bf hi = *(const v8bf*)(p + 16);
  v16bf f;
#pragma unroll
  for (int i = 0; i < 8; ++i) { f[i] = lo[i]; f[i + 8] = hi[i]; }
  return f;
}

static __device__ __forceinline__ v8f wmma_bf16(v16bf a, v16bf b, v8f c) {
  return __builtin_amdgcn_wmma_f32_16x16x32_bf16(false, a, false, b, (short)0, c,
                                                 false, false);
}

// ---- kernel 0: fp32 weights -> bf16 ---------------------------------------
__global__ void convert_weights_kernel(const float* __restrict__ wqkv,
                                       const float* __restrict__ wout,
                                       __bf16* __restrict__ wqkv_bf,
                                       __bf16* __restrict__ wout_bf) {
  int i = blockIdx.x * blockDim.x + threadIdx.x;
  if (i < 3 * CH * CH) wqkv_bf[i] = (__bf16)wqkv[i];
  if (i < CH * CH)     wout_bf[i] = (__bf16)wout[i];
}

// ---- kernel 1: GroupNorm, emit tokens [b, n, c] in bf16 -------------------
__global__ void __launch_bounds__(256) groupnorm_kernel(const float* __restrict__ x,
                                                        const float* __restrict__ gamma,
                                                        const float* __restrict__ beta,
                                                        __bf16* __restrict__ tokens) {
  const int b = blockIdx.x >> 5;       // 4 batches x 32 groups
  const int g = blockIdx.x & 31;
  const int c0 = g * CPG;
  const float* xb = x + (size_t)(b * CH + c0) * NTOK;   // 8 contiguous channel rows
  const int tid = threadIdx.x;

  float s = 0.f, ss = 0.f;
  for (int i = tid; i < CPG * NTOK; i += 256) {
    float v = xb[i];
    s += v; ss += v * v;
  }
#pragma unroll
  for (int off = 16; off > 0; off >>= 1) {
    s  += __shfl_down(s, off, 32);
    ss += __shfl_down(ss, off, 32);
  }
  __shared__ float red0[8], red1[8], stats[2];
  const int wv = tid >> 5;
  if ((tid & 31) == 0) { red0[wv] = s; red1[wv] = ss; }
  __syncthreads();
  if (tid == 0) {
    float t = 0.f, tt = 0.f;
    for (int i = 0; i < 8; ++i) { t += red0[i]; tt += red1[i]; }
    const float inv_n = 1.f / (float)(CPG * NTOK);
    float mean = t * inv_n;
    float var  = tt * inv_n - mean * mean;
    stats[0] = mean;
    stats[1] = rsqrtf(var + 1e-5f);
  }
  __syncthreads();
  const float mean = stats[0], rstd = stats[1];
  for (int i = tid; i < CPG * NTOK; i += 256) {
    int ch  = i >> 12;          // i / 4096
    int pix = i & (NTOK - 1);
    float v = (xb[i] - mean) * rstd * gamma[c0 + ch] + beta[c0 + ch];
    tokens[(size_t)(b * NTOK + pix) * CH + c0 + ch] = (__bf16)v;
  }
}

// ---- kernel 2: QKV projection GEMM: [16384,256] x [768,256]^T -------------
// Writes q,k row-major [b*n, c]; v transposed [b, c, n] for attention B-frags.
__global__ void __launch_bounds__(256) qkv_gemm_kernel(const __bf16* __restrict__ tok,
                                                       const __bf16* __restrict__ wq,
                                                       const float* __restrict__ bias,
                                                       __bf16* __restrict__ q,
                                                       __bf16* __restrict__ k,
                                                       __bf16* __restrict__ vt) {
  const int wave = threadIdx.x >> 5;
  const int lane = threadIdx.x & 31;
  const int row0 = blockIdx.x * 128 + wave * 16;   // token row (b folded)
  const int col0 = blockIdx.y * 64;                // output channel
  v8f acc[4] = {};
#pragma unroll
  for (int kc = 0; kc < 8; ++kc) {
    v16bf a = frag_a_rowmajor(tok, row0, CH, kc * 32);
#pragma unroll
    for (int j = 0; j < 4; ++j) {
      v16bf bf = frag_b_rowmajor(wq, col0 + j * 16, CH, kc * 32);
      acc[j] = wmma_bf16(a, bf, acc[j]);
    }
  }
  const int h = lane >> 4;
#pragma unroll
  for (int j = 0; j < 4; ++j) {
    int oc = col0 + j * 16 + (lane & 15);
    float bv = bias[oc];
#pragma unroll
    for (int r = 0; r < 8; ++r) {
      int row = row0 + r + 8 * h;
      __bf16 v = (__bf16)(acc[j][r] + bv);
      if (oc < CH) {
        q[(size_t)row * CH + oc] = v;
      } else if (oc < 2 * CH) {
        k[(size_t)row * CH + (oc - CH)] = v;
      } else {  // store V transposed: vt[b][c][n]
        vt[((size_t)(row >> 12) * CH + (oc - 2 * CH)) * NTOK + (row & (NTOK - 1))] = v;
      }
    }
  }
}

// ---- tile staging for attention: K tile (32x256) + V^T tile (256x32) ------
static __device__ __forceinline__ void stage_tiles(const __bf16* kb, const __bf16* vb,
                                                   int key0, __bf16* lk, __bf16* lv) {
  const int tid = threadIdx.x;
#if defined(USE_ASYNC_LDS)
  // Async DMA into LDS; tracked with ASYNCcnt (CDNA5 global_load_async_to_lds).
  const char* gk = (const char*)(kb + (size_t)key0 * CH);   // contiguous 16 KB
  char* dk = (char*)lk;
#pragma unroll
  for (int i = 0; i < 4; ++i)
    __builtin_amdgcn_global_load_async_to_lds_b128(
        (g_v4i*)(gk + (size_t)(tid + i * 256) * 16),
        (l_v4i*)(dk + (size_t)(tid + i * 256) * 16), 0, 0);
  const char* gv = (const char*)(vb + (size_t)tid * NTOK + key0);  // 64 B per channel row
  char* dv = (char*)(lv + tid * 32);
#pragma unroll
  for (int i = 0; i < 4; ++i)
    __builtin_amdgcn_global_load_async_to_lds_b128(
        (g_v4i*)(gv + i * 16), (l_v4i*)(dv + i * 16), 0, 0);
#else
  const uint4* src = (const uint4*)(kb + (size_t)key0 * CH);
  uint4* dst = (uint4*)lk;
#pragma unroll
  for (int i = 0; i < 4; ++i) dst[tid + i * 256] = src[tid + i * 256];
  const uint4* vs = (const uint4*)(vb + (size_t)tid * NTOK + key0);
  uint4* vd = (uint4*)(lv + tid * 32);
#pragma unroll
  for (int i = 0; i < 4; ++i) vd[i] = vs[i];
#endif
}

static __device__ __forceinline__ void wait_stage() {
#if defined(USE_ASYNC_LDS)
  __builtin_amdgcn_s_wait_asynccnt(0);
#endif
}

// ---- kernel 3: flash attention --------------------------------------------
// Per block: 128 query rows (8 waves x 16), batch = blockIdx.y.
// Computes S^T = K*Q^T so the exp'd scores drop straight into the P A-fragment.
__global__ void __launch_bounds__(256, 1) attn_kernel(const __bf16* __restrict__ qg,
                                                      const __bf16* __restrict__ kg,
                                                      const __bf16* __restrict__ vtg,
                                                      __bf16* __restrict__ og) {
  __shared__ __bf16 lds_k[2][32 * CH];     // 2 x 16 KB (32 keys x 256 ch, row-major)
  __shared__ __bf16 lds_vt[2][CH * 32];    // 2 x 16 KB (256 ch x 32 keys, ch-major)

  const int batch = blockIdx.y;
  const int wave  = threadIdx.x >> 5;
  const int lane  = threadIdx.x & 31;
  const int h     = lane >> 4;
  const int qrow0 = blockIdx.x * 128 + wave * 16;

  const __bf16* qb = qg  + (size_t)batch * NC;
  const __bf16* kb = kg  + (size_t)batch * NC;
  const __bf16* vb = vtg + (size_t)batch * NC;

  // Q fragments (B operand of S^T): 16 query cols x K=256, kept resident
  v16bf qf[8];
#pragma unroll
  for (int kc = 0; kc < 8; ++kc) qf[kc] = frag_b_rowmajor(qb, qrow0, CH, kc * 32);

  v8f oacc[16] = {};                 // 16 query rows x 256 channels, fp32
  float row_m = -1e30f, row_l = 0.f; // this lane owns query row qrow0 + (lane&15)
  const float scale = 0.0625f;       // 256^-0.5

  // prologue: stage tile 0
  stage_tiles(kb, vb, 0, lds_k[0], lds_vt[0]);
  wait_stage();
  __syncthreads();

  for (int t = 0; t < NTOK / 32; ++t) {
    const int cur = t & 1;
    // stage next tile into the other buffer while we compute on this one
    if (t + 1 < NTOK / 32) {
      stage_tiles(kb, vb, (t + 1) * 32, lds_k[1 - cur], lds_vt[1 - cur]);
      if (t + 2 < NTOK / 32)
        __builtin_prefetch(kb + (size_t)(t + 2) * 32 * CH, 0, 1);  // global_prefetch_b8
    }
    const __bf16* Lk = lds_k[cur];
    const __bf16* Lv = lds_vt[cur];

    // S^T = K * Q^T : two 16-key tiles, K-dim = 256 in 8 WMMA steps
    v8f st0 = {}, st1 = {};
#pragma unroll
    for (int kc = 0; kc < 8; ++kc) {
      v16bf a0, a1;
      {
        const __bf16* p0 = &Lk[(size_t)(lane & 15) * CH + kc * 32 + h * 8];
        const __bf16* p1 = p0 + 16 * CH;
        v8bf lo0 = *(const v8bf*)p0;  v8bf hi0 = *(const v8bf*)(p0 + 16);
        v8bf lo1 = *(const v8bf*)p1;  v8bf hi1 = *(const v8bf*)(p1 + 16);
#pragma unroll
        for (int i = 0; i < 8; ++i) {
          a0[i] = lo0[i]; a0[i + 8] = hi0[i];
          a1[i] = lo1[i]; a1[i + 8] = hi1[i];
        }
      }
      st0 = wmma_bf16(a0, qf[kc], st0);
      st1 = wmma_bf16(a1, qf[kc], st1);
    }

    // online softmax: each lane holds 16 of this row's 32 scores
    float mloc = -1e30f;
#pragma unroll
    for (int i = 0; i < 8; ++i) {
      st0[i] *= scale; st1[i] *= scale;
      mloc = fmaxf(mloc, fmaxf(st0[i], st1[i]));
    }
    mloc = fmaxf(mloc, __shfl_xor(mloc, 16, 32));
    float m_new = fmaxf(row_m, mloc);
    float alpha = __expf(row_m - m_new);

    // P fragment: D-of-S^T maps 1:1 onto bf16 A-fragment layout (no shuffles)
    v16bf pf;
    float rs = 0.f;
#pragma unroll
    for (int i = 0; i < 8; ++i) {
      float e0 = __expf(st0[i] - m_new);
      float e1 = __expf(st1[i] - m_new);
      rs += e0 + e1;
      pf[i]     = (__bf16)e0;
      pf[i + 8] = (__bf16)e1;
    }
    rs += __shfl_xor(rs, 16, 32);
    row_l = row_l * alpha + rs;
    row_m = m_new;

    // rescale O: O-tile row (VGPR r, half h) needs alpha of query row r + 8h
    float fac[8];
#pragma unroll
    for (int r = 0; r < 8; ++r) fac[r] = __shfl(alpha, h * 8 + r, 32);
#pragma unroll
    for (int ct = 0; ct < 16; ++ct)
#pragma unroll
      for (int r = 0; r < 8; ++r) oacc[ct][r] *= fac[r];

    // O += P(16x32) * V(32x16) per channel tile
#pragma unroll
    for (int ct = 0; ct < 16; ++ct) {
      const __bf16* p = &Lv[(size_t)(ct * 16 + (lane & 15)) * 32 + h * 8];
      v8bf vlo = *(const v8bf*)p;  v8bf vhi = *(const v8bf*)(p + 16);
      v16bf bf;
#pragma unroll
      for (int i = 0; i < 8; ++i) { bf[i] = vlo[i]; bf[i + 8] = vhi[i]; }
      oacc[ct] = wmma_bf16(pf, bf, oacc[ct]);
    }

    // my async loads for t+1 must land, then all waves sync before buffer swap
    wait_stage();
    __syncthreads();
  }

  // normalize and store [b, n, c] bf16
  float inv = 1.f / row_l;
  float fac[8];
#pragma unroll
  for (int r = 0; r < 8; ++r) fac[r] = __shfl(inv, h * 8 + r, 32);
  __bf16* ob = og + (size_t)batch * NC;
#pragma unroll
  for (int ct = 0; ct < 16; ++ct) {
    int cc = ct * 16 + (lane & 15);
#pragma unroll
    for (int r = 0; r < 8; ++r) {
      int row = qrow0 + r + 8 * h;
      ob[(size_t)row * CH + cc] = (__bf16)(oacc[ct][r] * fac[r]);
    }
  }
}

// ---- kernel 4: out projection + bias + residual, f32 output [b,c,h,w] -----
__global__ void __launch_bounds__(256) out_proj_kernel(const __bf16* __restrict__ attn,
                                                       const __bf16* __restrict__ wo,
                                                       const float* __restrict__ bias,
                                                       const float* __restrict__ x,
                                                       float* __restrict__ out) {
  const int wave = threadIdx.x >> 5;
  const int lane = threadIdx.x & 31;
  const int row0 = blockIdx.x * 128 + wave * 16;
  const int col0 = blockIdx.y * 64;
  v8f acc[4] = {};
#pragma unroll
  for (int kc = 0; kc < 8; ++kc) {
    v16bf a = frag_a_rowmajor(attn, row0, CH, kc * 32);
#pragma unroll
    for (int j = 0; j < 4; ++j) {
      v16bf bf = frag_b_rowmajor(wo, col0 + j * 16, CH, kc * 32);
      acc[j] = wmma_bf16(a, bf, acc[j]);
    }
  }
  const int h = lane >> 4;
#pragma unroll
  for (int j = 0; j < 4; ++j) {
    int oc = col0 + j * 16 + (lane & 15);
    float bv = bias[oc];
#pragma unroll
    for (int r = 0; r < 8; ++r) {
      int row = row0 + r + 8 * h;
      size_t idx = ((size_t)(row >> 12) * CH + oc) * NTOK + (row & (NTOK - 1));
      out[idx] = acc[j][r] + bv + x[idx];
    }
  }
}

// ---------------------------------------------------------------------------
extern "C" void kernel_launch(void* const* d_in, const int* in_sizes, int n_in,
                              void* d_out, int out_size, void* d_ws, size_t ws_size,
                              hipStream_t stream) {
  (void)in_sizes; (void)n_in; (void)out_size; (void)ws_size;
  const float* x     = (const float*)d_in[0];
  const float* gamma = (const float*)d_in[1];
  const float* beta  = (const float*)d_in[2];
  const float* w_qkv = (const float*)d_in[3];
  const float* b_qkv = (const float*)d_in[4];
  const float* w_out = (const float*)d_in[5];
  const float* b_out = (const float*)d_in[6];

  char* ws = (char*)d_ws;
  size_t off = 0;
  auto alloc = [&](size_t bytes) -> char* {
    char* p = ws + off;
    off += (bytes + 255) & ~(size_t)255;
    return p;
  };
  __bf16* wq_bf = (__bf16*)alloc((size_t)3 * CH * CH * 2);   // 384 KB
  __bf16* wo_bf = (__bf16*)alloc((size_t)CH * CH * 2);       // 128 KB
  __bf16* tok   = (__bf16*)alloc((size_t)BATCH * NC * 2);    // 8 MB
  __bf16* q     = (__bf16*)alloc((size_t)BATCH * NC * 2);
  __bf16* k     = (__bf16*)alloc((size_t)BATCH * NC * 2);
  __bf16* vt    = (__bf16*)alloc((size_t)BATCH * NC * 2);
  __bf16* at    = (__bf16*)alloc((size_t)BATCH * NC * 2);

  convert_weights_kernel<<<(3 * CH * CH + 255) / 256, 256, 0, stream>>>(
      w_qkv, w_out, wq_bf, wo_bf);
  groupnorm_kernel<<<BATCH * GROUPS, 256, 0, stream>>>(x, gamma, beta, tok);
  qkv_gemm_kernel<<<dim3((BATCH * NTOK) / 128, (3 * CH) / 64), 256, 0, stream>>>(
      tok, wq_bf, b_qkv, q, k, vt);
  attn_kernel<<<dim3(NTOK / 128, BATCH), 256, 0, stream>>>(q, k, vt, at);
  out_proj_kernel<<<dim3((BATCH * NTOK) / 128, CH / 64), 256, 0, stream>>>(
      at, wo_bf, b_out, x, (float*)d_out);
}
